// RGCN_37555194036548
// MI455X (gfx1250) — matrix operand order repaired and checked by reading, not compiled, not run on hardware
//
#include <hip/hip_runtime.h>
#include <hip/hip_bf16.h>
#include <stddef.h>

// ---------------------------------------------------------------------------
// RGCN on gfx1250: bf16 WMMA GEMMs + native fp32 atomic scatter.
// Dims: N=50000 (=16*3125), E=1.6M, D=128, R=16, B=4, C=16. All /16 exact.
// ---------------------------------------------------------------------------

typedef __attribute__((ext_vector_type(16))) __bf16 v16bf;
typedef __attribute__((ext_vector_type(8)))  __bf16 v8bf;
typedef __attribute__((ext_vector_type(4)))  __bf16 v4bf;
typedef __attribute__((ext_vector_type(8)))  float  v8f;
typedef __attribute__((ext_vector_type(4)))  float  v4f;

#define LDS_STRIDE 136  // 128 + 8 halfs pad: rows land on distinct bank groups

// ---- tiny prep kernels -----------------------------------------------------

// Wt[r][o][k] = sum_b c[r,b] * wb[b][k][o]   (bf16, pre-transposed weight)
template <int DOUT>
__global__ void basis_combine(const float* __restrict__ wb,
                              const float* __restrict__ c,
                              __bf16* __restrict__ Wt) {
  int idx = blockIdx.x * blockDim.x + threadIdx.x;
  const int total = 16 * DOUT * 128;
  if (idx >= total) return;
  int r = idx / (DOUT * 128);
  int rem = idx - r * (DOUT * 128);
  int o = rem / 128;
  int k = rem - o * 128;
  float v = 0.f;
#pragma unroll
  for (int b = 0; b < 4; ++b)
    v += c[r * 4 + b] * wb[((size_t)b * 128 + k) * DOUT + o];
  Wt[((size_t)r * DOUT + o) * 128 + k] = (__bf16)v;
}

// lwt[o][k] = lw[k][o]  (bf16 transpose of self-loop weight)
template <int DOUT>
__global__ void transpose_lw(const float* __restrict__ lw,
                             __bf16* __restrict__ lwt) {
  int idx = blockIdx.x * blockDim.x + threadIdx.x;
  if (idx >= DOUT * 128) return;
  int o = idx / 128;
  int k = idx - o * 128;
  lwt[o * 128 + k] = (__bf16)lw[(size_t)k * DOUT + o];
}

// 4 elems/thread: b128 load -> packed cvt -> b64 store
__global__ void f32_to_bf16_v4(const float* __restrict__ in,
                               __bf16* __restrict__ out, int n4) {
  int idx = blockIdx.x * blockDim.x + threadIdx.x;
  if (idx >= n4) return;
  v4f v = *(const v4f*)(in + (size_t)idx * 4);
  v4bf o = {(__bf16)v.x, (__bf16)v.y, (__bf16)v.z, (__bf16)v.w};
  *(v4bf*)(out + (size_t)idx * 4) = o;
}

__global__ void relu_to_bf16_v4(const float* __restrict__ in,
                                __bf16* __restrict__ out, int n4) {
  int idx = blockIdx.x * blockDim.x + threadIdx.x;
  if (idx >= n4) return;
  v4f v = *(const v4f*)(in + (size_t)idx * 4);
  v4bf o = {(__bf16)fmaxf(v.x, 0.f), (__bf16)fmaxf(v.y, 0.f),
            (__bf16)fmaxf(v.z, 0.f), (__bf16)fmaxf(v.w, 0.f)};
  *(v4bf*)(out + (size_t)idx * 4) = o;
}

// ---- WMMA GEMM -------------------------------------------------------------
// Block = (DOUT/16) waves. Grid = (Nn/16, nRel). Computes, for relation
// r = blockIdx.y:  out[m][n] = sum_k A[m][k] * Wt[r][n][k]   (K = 128)
// BF16OUT: store bf16 into xw[r][m][n]; else fp32 out[m][n] + bias[n].
template <int DOUT, bool BF16OUT>
__global__ __launch_bounds__(32 * (DOUT / 16)) void gemm_wmma(
    const __bf16* __restrict__ A, const __bf16* __restrict__ Wt,
    void* __restrict__ out, const float* __restrict__ bias, int Nn) {
  constexpr int NTILES = DOUT / 16;
  __shared__ __bf16 lds[16 * LDS_STRIDE];

  const int tileM = blockIdx.x;
  const int r = blockIdx.y;
  const __bf16* Wr = Wt + (size_t)r * DOUT * 128;
  const int tid = threadIdx.x;

  // Stage 16x128 A tile into LDS (16B chunks).
  const __bf16* Arow = A + (size_t)tileM * 16 * 128;
  for (int i = tid; i < 16 * 16; i += 32 * NTILES) {
    int row = i >> 4;
    int col = (i & 15) << 3;
    *(v8bf*)&lds[row * LDS_STRIDE + col] = *(const v8bf*)&Arow[row * 128 + col];
  }
  __syncthreads();

  const int wave = tid >> 5;
  const int lane = tid & 31;
  const int lrow = lane & 15;
  const int hi = lane >> 4;  // half-wave select
  const int tileN = wave * 16;

  v8f acc = {};
#pragma unroll
  for (int s = 0; s < 4; ++s) {
    const int kb = s * 32;
    // A fragment (ISA 16-bit 16x32 layout): lanes 0-15 -> K {kb..kb+7, kb+16..23},
    // lanes 16-31 -> K {kb+8..15, kb+24..31}; row = lane&15.
    const __bf16* ap = &lds[lrow * LDS_STRIDE + kb + hi * 8];
    v8bf a0 = *(const v8bf*)ap;
    v8bf a1 = *(const v8bf*)(ap + 16);
    v16bf a = __builtin_shufflevector(a0, a1, 0, 1, 2, 3, 4, 5, 6, 7, 8, 9, 10,
                                      11, 12, 13, 14, 15);
    // B fragment: lane n holds column n; same K order as A: lanes 0-15 K {kb..kb+7, kb+16..23},
    // lanes 16-31 K {kb+8..15, kb+24..31}. Wt is pre-transposed -> two 16B loads per lane.
    const __bf16* bp = Wr + (size_t)(tileN + lrow) * 128 + kb + hi * 8;
    v8bf b0 = *(const v8bf*)bp;
    v8bf b1 = *(const v8bf*)(bp + 16);
    v16bf b = __builtin_shufflevector(b0, b1, 0, 1, 2, 3, 4, 5, 6, 7, 8, 9, 10,
                                      11, 12, 13, 14, 15);
    acc = __builtin_amdgcn_wmma_f32_16x16x32_bf16(false, a, false, b, (short)0,
                                                  acc, false, false);
  }

  // D layout: VGPR j, lanes 0-15 -> (M=j, N=lane); lanes 16-31 -> (M=j+8, N=lane-16)
  const int n = tileN + lrow;
  if constexpr (BF16OUT) {
    __bf16* O = (__bf16*)out + ((size_t)r * Nn + (size_t)tileM * 16) * DOUT;
#pragma unroll
    for (int j = 0; j < 8; ++j)
      O[(size_t)(j + hi * 8) * DOUT + n] = (__bf16)acc[j];
  } else {
    float* O = (float*)out + (size_t)tileM * 16 * DOUT;
    const float bv = bias[n];
#pragma unroll
    for (int j = 0; j < 8; ++j)
      O[(size_t)(j + hi * 8) * DOUT + n] = acc[j] + bv;
  }
}

// ---- edge gather + atomic scatter -----------------------------------------
// One thread per 4-element chunk of one edge's message.
// unsafeAtomicAdd -> non-returning global_atomic_add_f32 (agg fits in L2).
template <int DOUT>
__global__ void edge_scatter(const __bf16* __restrict__ xw,
                             const int* __restrict__ src,
                             const int* __restrict__ dst,
                             const int* __restrict__ etype,
                             const float* __restrict__ norm,
                             float* __restrict__ agg, int E, int Nn) {
  constexpr int CPE = DOUT / 4;
  int gid = blockIdx.x * blockDim.x + threadIdx.x;
  int e = gid / CPE;
  if (e >= E) return;
  int c = (gid - e * CPE) * 4;
  const int s = src[e];
  const int t = etype[e];
  const int d = dst[e];
  const float nv = norm[e];
  const __bf16* p = xw + ((size_t)t * Nn + s) * DOUT + c;
  v4bf m = *(const v4bf*)p;
  float* o = agg + (size_t)d * DOUT + c;
#pragma unroll
  for (int i = 0; i < 4; ++i)
    unsafeAtomicAdd(&o[i], (float)m[i] * nv);
}

// ---------------------------------------------------------------------------

static inline size_t align256(size_t x) { return (x + 255) & ~(size_t)255; }

extern "C" void kernel_launch(void* const* d_in, const int* in_sizes, int n_in,
                              void* d_out, int out_size, void* d_ws,
                              size_t ws_size, hipStream_t stream) {
  const float* x = (const float*)d_in[0];
  const int* src = (const int*)d_in[1];
  const int* dst = (const int*)d_in[2];
  const int* ety = (const int*)d_in[3];
  const float* norm = (const float*)d_in[4];
  const float* wb[3] = {(const float*)d_in[5], (const float*)d_in[9],
                        (const float*)d_in[13]};
  const float* cc[3] = {(const float*)d_in[6], (const float*)d_in[10],
                        (const float*)d_in[14]};
  const float* lw[3] = {(const float*)d_in[7], (const float*)d_in[11],
                        (const float*)d_in[15]};
  const float* bi[3] = {(const float*)d_in[8], (const float*)d_in[12],
                        (const float*)d_in[16]};

  const int Nn = in_sizes[0] / 128;  // 50000
  const int E = in_sizes[1];         // 1600000
  const int Mtiles = Nn / 16;        // 3125

  // workspace partition
  char* w = (char*)d_ws;
  __bf16* hb = (__bf16*)w;  w += align256((size_t)Nn * 128 * 2);
  __bf16* Wt = (__bf16*)w;  w += align256((size_t)16 * 128 * 128 * 2);
  __bf16* lwt = (__bf16*)w; w += align256((size_t)128 * 128 * 2);
  __bf16* xw = (__bf16*)w;  w += align256((size_t)16 * Nn * 128 * 2);
  float* agg = (float*)w;   w += align256((size_t)Nn * 128 * 4);

  // input -> bf16
  {
    int n4 = Nn * 128 / 4;
    f32_to_bf16_v4<<<(n4 + 255) / 256, 256, 0, stream>>>(x, hb, n4);
  }

  // ---- layers 0 and 1 (D=128 -> 128, ReLU) ----
  for (int l = 0; l < 2; ++l) {
    basis_combine<128><<<(16 * 128 * 128 + 255) / 256, 256, 0, stream>>>(
        wb[l], cc[l], Wt);
    transpose_lw<128><<<(128 * 128 + 255) / 256, 256, 0, stream>>>(lw[l], lwt);
    // xw[r] = hb @ W[r]  (bf16 out)
    gemm_wmma<128, true><<<dim3(Mtiles, 16), 256, 0, stream>>>(hb, Wt, xw,
                                                               nullptr, Nn);
    // agg = hb @ lw + b  (fp32, fully initializes agg)
    gemm_wmma<128, false><<<dim3(Mtiles, 1), 256, 0, stream>>>(hb, lwt, agg,
                                                               bi[l], Nn);
    // agg += scatter_sum(xw[etype, src] * norm)
    {
      long threads = (long)E * 32;
      edge_scatter<128><<<(threads + 255) / 256, 256, 0, stream>>>(
          xw, src, dst, ety, norm, agg, E, Nn);
    }
    int n4 = Nn * 128 / 4;
    relu_to_bf16_v4<<<(n4 + 255) / 256, 256, 0, stream>>>(agg, hb, n4);
  }

  // ---- layer 2 (D=128 -> 16, no activation, fp32 out) ----
  {
    float* out = (float*)d_out;
    basis_combine<16><<<(16 * 16 * 128 + 255) / 256, 256, 0, stream>>>(
        wb[2], cc[2], Wt);
    transpose_lw<16><<<(16 * 128 + 255) / 256, 256, 0, stream>>>(lw[2], lwt);
    gemm_wmma<16, true><<<dim3(Mtiles, 16), 32, 0, stream>>>(hb, Wt, xw,
                                                             nullptr, Nn);
    gemm_wmma<16, false><<<dim3(Mtiles, 1), 32, 0, stream>>>(hb, lwt, out,
                                                             bi[2], Nn);
    long threads = (long)E * 4;
    edge_scatter<16><<<(threads + 255) / 256, 256, 0, stream>>>(
        xw, src, dst, ety, norm, out, E, Nn);
  }
}
